// FAUKernel_thw_50972671869073
// MI455X (gfx1250) — matrix operand order repaired and checked by reading, not compiled, not run on hardware
//
#include <hip/hip_runtime.h>
#include <hip/hip_bf16.h>

typedef __attribute__((ext_vector_type(16))) __bf16 v16bf;
typedef __attribute__((ext_vector_type(8)))  float  v8f;

union Frag { v16bf v; unsigned u[8]; };

#define LDSTRIDE 34   // 32 + 2 pad (uint16 units); rows at 68B -> 4B aligned pair reads

__device__ __forceinline__ unsigned short f2bf(float f) {
  unsigned u = __float_as_uint(f);
  u += 0x7FFFu + ((u >> 16) & 1u);      // round-to-nearest-even
  return (unsigned short)(u >> 16);
}
__device__ __forceinline__ unsigned f2bf2(float lo, float hi) {
  return (unsigned)f2bf(lo) | ((unsigned)f2bf(hi) << 16);
}
__device__ __forceinline__ float bf2f(unsigned short h) {
  return __uint_as_float(((unsigned)h) << 16);
}

// A fragment: 16x32 bf16, LDS row-major [m][k]. ISA layout: lane m=lane&15,
// h=lane>>4; VGPR v holds K pair k0=(v>>2)*16 + h*8 + (v&3)*2.
__device__ __forceinline__ void loadA(Frag& a, const unsigned short* As, int mBase, int lane) {
  const int m = mBase + (lane & 15);
  const int h = lane >> 4;
  const unsigned short* row = As + m * LDSTRIDE;
#pragma unroll
  for (int v = 0; v < 8; ++v) {
    int k0 = ((v >> 2) << 4) + (h << 3) + ((v & 3) << 1);
    a.u[v] = *(const unsigned*)(row + k0);
  }
}

// B fragment: 32x16 bf16, LDS staged K-major per column: [n][k].
// ISA layout: lane n=lane&15, h=lane>>4; VGPR v holds K pair k0=(v>>2)*16 + h*8 + (v&3)*2.
__device__ __forceinline__ void loadB(Frag& b, const unsigned short* Bs, int nBase, int lane) {
  const int n = nBase + (lane & 15);
  const int h = lane >> 4;
  const unsigned short* row = Bs + n * LDSTRIDE;
#pragma unroll
  for (int v = 0; v < 8; ++v) {
    int k0 = ((v >> 2) << 4) + (h << 3) + ((v & 3) << 1);
    b.u[v] = *(const unsigned*)(row + k0);
  }
}

// ---------------------------------------------------------------------------
// Kernel 1/4: Y[b,l,n] = ReLU(scale[l] * sum_c W[l,c]*X[b,c,n] + shift[l])
// M=64 (all L), Ntile=128, K=C=512. Output bf16.
// Register double-buffered: next tile's global loads issued while WMMAs run.
// ---------------------------------------------------------------------------
__global__ __launch_bounds__(256)
void phi_gemm_kernel(const float* __restrict__ X,      // (B,512,6272)
                     const float* __restrict__ W,      // (64,512)
                     const float* __restrict__ gamma,
                     const float* __restrict__ beta,
                     const float* __restrict__ mean,
                     const float* __restrict__ var,
                     unsigned short* __restrict__ Ybf) // (B,64,6272)
{
  __shared__ unsigned short As[64 * LDSTRIDE];
  __shared__ unsigned short Bs[128 * LDSTRIDE];
  __shared__ float scl[64], shf[64];

  const int t  = threadIdx.x;
  const int b  = blockIdx.y;
  const int n0 = blockIdx.x * 128;

  if (t < 64) {
    float s = gamma[t] * rsqrtf(var[t] + 1e-5f);
    scl[t] = s;
    shf[t] = beta[t] - mean[t] * s;
  }

  const int lane  = t & 31;
  const int wave  = t >> 5;
  const int mBase = (wave >> 1) << 4;   // 0,16,32,48
  const int nHalf = (wave & 1) << 6;    // 0 or 64
  v8f acc[4] = {};

  const float* Xb = X + (size_t)b * 512 * 6272;

  float2 aReg[4];
  float  bReg[16];
  auto loadTile = [&](int kt) {
    const int c0 = kt * 32;
#pragma unroll
    for (int i = 0; i < 4; ++i) {            // A: W tile 64x32, paired (K-contig)
      int idx = t + i * 256;                 // 1024 float2's
      aReg[i] = *(const float2*)&W[(idx >> 4) * 512 + c0 + ((idx & 15) << 1)];
    }
#pragma unroll
    for (int i = 0; i < 16; ++i) {           // B: X tile 32(c) x 128(n), elementwise
      int idx = t + i * 256;
      bReg[i] = Xb[(c0 + (idx >> 7)) * 6272 + n0 + (idx & 127)];
    }
  };

  loadTile(0);
  for (int kt = 0; kt < 16; ++kt) {
#pragma unroll
    for (int i = 0; i < 4; ++i) {
      int idx = t + i * 256;
      *(unsigned*)&As[(idx >> 4) * LDSTRIDE + ((idx & 15) << 1)] = f2bf2(aReg[i].x, aReg[i].y);
    }
#pragma unroll
    for (int i = 0; i < 16; ++i) {
      int idx = t + i * 256;
      Bs[(idx & 127) * LDSTRIDE + (idx >> 7)] = f2bf(bReg[i]);
    }
    __syncthreads();
    if (kt < 15) loadTile(kt + 1);           // overlap global latency with WMMA
    Frag a, bf_;
    loadA(a, As, mBase, lane);
#pragma unroll
    for (int j = 0; j < 4; ++j) {
      loadB(bf_, Bs, nHalf + j * 16, lane);
      acc[j] = __builtin_amdgcn_wmma_f32_16x16x32_bf16(false, a.v, false, bf_.v,
                                                       (short)0, acc[j], false, false);
    }
    __syncthreads();
  }

  const int h = lane >> 4, nc = lane & 15;
#pragma unroll
  for (int j = 0; j < 4; ++j) {
    int n = n0 + nHalf + j * 16 + nc;
#pragma unroll
    for (int r = 0; r < 8; ++r) {
      int l = mBase + r + (h << 3);
      float y = fmaxf(acc[j][r] * scl[l] + shf[l], 0.0f);
      Ybf[((size_t)b * 64 + l) * 6272 + n] = f2bf(y);
    }
  }
}

// ---------------------------------------------------------------------------
// Row L2 scale: rscale[b*64+l] = 1/max(||Y1[b,l,:]||, 1e-12)
// ---------------------------------------------------------------------------
__global__ __launch_bounds__(256)
void row_scale_kernel(const unsigned short* __restrict__ Ybf, float* __restrict__ rscale) {
  __shared__ float red[256];
  const int row = blockIdx.x;
  const unsigned short* p = Ybf + (size_t)row * 6272;
  float s = 0.f;
  for (int n = threadIdx.x; n < 6272; n += 256) {
    float v = bf2f(p[n]);
    s += v * v;
  }
  red[threadIdx.x] = s;
  __syncthreads();
  for (int off = 128; off > 0; off >>= 1) {
    if (threadIdx.x < off) red[threadIdx.x] += red[threadIdx.x + off];
    __syncthreads();
  }
  if (threadIdx.x == 0) rscale[row] = 1.0f / fmaxf(sqrtf(red[0]), 1e-12f);
}

// ---------------------------------------------------------------------------
// Kernel 2 (split-K): latPart[p,b,l,c] = sum_{n in chunk p} Y1[b,l,n]*K[b,c,n]
// M=64 (l), Ntile=128 (c), K=6272 split into 14 chunks of 14 k-iters (448 n).
// 448 workgroups -> enough in-flight loads to stream key at HBM rate.
// Both tiles are K-contiguous in memory -> fully paired (b64/b32) staging.
// ---------------------------------------------------------------------------
__global__ __launch_bounds__(256)
void latent_gemm_kernel(const unsigned short* __restrict__ Y1bf, // (B,64,6272)
                        const float* __restrict__ K,             // (B,512,6272)
                        float* __restrict__ latPart)             // (14,B,64,512)
{
  __shared__ unsigned short As[64 * LDSTRIDE];
  __shared__ unsigned short Bs[128 * LDSTRIDE];

  const int t     = threadIdx.x;
  const int c0    = blockIdx.x * 128;
  const int chunk = blockIdx.y;
  const int b     = blockIdx.z;
  const int kt0   = chunk * 14;

  const int lane  = t & 31;
  const int wave  = t >> 5;
  const int mBase = (wave >> 1) << 4;
  const int nHalf = (wave & 1) << 6;
  v8f acc[4] = {};

  const unsigned short* Yb = Y1bf + (size_t)b * 64 * 6272;
  const float*          Kb = K    + (size_t)b * 512 * 6272;

  unsigned aReg[4];      // A: 2 bf16 per dword (raw copy, no conversion)
  float2   bReg[8];      // B: paired float loads
  auto loadTile = [&](int kt) {
    const int kn0 = kt * 32;
#pragma unroll
    for (int i = 0; i < 4; ++i) {            // A: Y1 tile 64x32 bf16, 1024 dwords
      int idx = t + i * 256;
      aReg[i] = *(const unsigned*)&Yb[(idx >> 4) * 6272 + kn0 + ((idx & 15) << 1)];
    }
#pragma unroll
    for (int i = 0; i < 8; ++i) {            // B: key tile 128(c) x 32(k), 2048 float2's
      int idx = t + i * 256;
      bReg[i] = *(const float2*)&Kb[(c0 + (idx >> 4)) * 6272 + kn0 + ((idx & 15) << 1)];
    }
  };

  loadTile(kt0);
  for (int kk = 0; kk < 14; ++kk) {
#pragma unroll
    for (int i = 0; i < 4; ++i) {
      int idx = t + i * 256;
      *(unsigned*)&As[(idx >> 4) * LDSTRIDE + ((idx & 15) << 1)] = aReg[i];
    }
#pragma unroll
    for (int i = 0; i < 8; ++i) {
      int idx = t + i * 256;
      *(unsigned*)&Bs[(idx >> 4) * LDSTRIDE + ((idx & 15) << 1)] = f2bf2(bReg[i].x, bReg[i].y);
    }
    __syncthreads();
    if (kk < 13) loadTile(kt0 + kk + 1);
    Frag a, bf_;
    loadA(a, As, mBase, lane);
#pragma unroll
    for (int j = 0; j < 4; ++j) {
      loadB(bf_, Bs, nHalf + j * 16, lane);
      acc[j] = __builtin_amdgcn_wmma_f32_16x16x32_bf16(false, a.v, false, bf_.v,
                                                       (short)0, acc[j], false, false);
    }
    __syncthreads();
  }

  const int h = lane >> 4, nc = lane & 15;
  float* dst = latPart + ((size_t)chunk * 8 + b) * 64 * 512;
#pragma unroll
  for (int j = 0; j < 4; ++j) {
    int c = c0 + nHalf + j * 16 + nc;
#pragma unroll
    for (int r = 0; r < 8; ++r) {
      int l = mBase + r + (h << 3);
      dst[l * 512 + c] = acc[j][r];
    }
  }
}

// Fixed-order split-K reduction (deterministic), folds per-row L2 scale.
__global__ __launch_bounds__(256)
void latent_reduce_kernel(const float* __restrict__ latPart, // (14,B,64,512)
                          const float* __restrict__ rscale,  // (B*64)
                          float* __restrict__ latent)        // (B,64,512)
{
  int i = blockIdx.x * 256 + threadIdx.x;    // over 8*64*512 = 262144
  if (i >= 262144) return;
  float s = 0.f;
#pragma unroll
  for (int p = 0; p < 14; ++p) s += latPart[(size_t)p * 262144 + i];
  latent[i] = s * rscale[i >> 9];            // i>>9 == b*64 + l
}

// ---------------------------------------------------------------------------
// Kernel 3 (tiny, per batch): ln = l2norm(latent); aff = softmax(ln ln^T);
// latent2 = aff @ latent, stored transposed bf16: latT[b,c,l].
// ---------------------------------------------------------------------------
__global__ __launch_bounds__(256)
void affinity_kernel(const float* __restrict__ latent,   // (B,64,512)
                     unsigned short* __restrict__ latT)  // (B,512,64) bf16
{
  __shared__ float S[64 * 64];
  __shared__ float rinv[64];
  const int t = threadIdx.x;
  const int b = blockIdx.x;
  const float* Lb = latent + (size_t)b * 64 * 512;

  if (t < 64) {
    float s = 0.f;
    const float* row = Lb + t * 512;
    for (int c = 0; c < 512; ++c) { float v = row[c]; s += v * v; }
    rinv[t] = 1.0f / fmaxf(sqrtf(s), 1e-12f);
  }
  __syncthreads();

  for (int i = t; i < 4096; i += 256) {
    int l = i >> 6, m = i & 63;
    const float* rl = Lb + l * 512;
    const float* rm = Lb + m * 512;
    float s = 0.f;
    for (int c = 0; c < 512; ++c) s += rl[c] * rm[c];
    S[i] = s * rinv[l] * rinv[m];
  }
  __syncthreads();

  if (t < 64) {                               // softmax over m (last axis)
    float* row = S + t * 64;
    float mx = row[0];
    for (int m = 1; m < 64; ++m) mx = fmaxf(mx, row[m]);
    float sum = 0.f;
    for (int m = 0; m < 64; ++m) { float e = __expf(row[m] - mx); row[m] = e; sum += e; }
    float inv = 1.0f / sum;
    for (int m = 0; m < 64; ++m) row[m] *= inv;
  }
  __syncthreads();

  for (int i = t; i < 32768; i += 256) {      // latent2[l][c], write latT[c][l] bf16
    int c = i >> 6, l = i & 63;
    const float* arow = S + l * 64;
    float s = 0.f;
    for (int m = 0; m < 64; ++m) s += arow[m] * Lb[m * 512 + c];
    latT[((size_t)b * 512 + c) * 64 + l] = f2bf(s);
  }
}

// ---------------------------------------------------------------------------
// Column L2 scale over L: cscale[b*6272+n] = 1/max(||Y2[b,:,n]||, 1e-12)
// ---------------------------------------------------------------------------
__global__ __launch_bounds__(256)
void col_scale_kernel(const unsigned short* __restrict__ Y2bf, float* __restrict__ cscale) {
  int tid = blockIdx.x * 256 + threadIdx.x;
  if (tid >= 8 * 6272) return;
  int b = tid / 6272, n = tid - b * 6272;
  const unsigned short* p = Y2bf + (size_t)b * 64 * 6272 + n;
  float s = 0.f;
#pragma unroll 8
  for (int l = 0; l < 64; ++l) { float v = bf2f(p[(size_t)l * 6272]); s += v * v; }
  cscale[tid] = 1.0f / fmaxf(sqrtf(s), 1e-12f);
}

// ---------------------------------------------------------------------------
// Kernel 5: out[b,c,n] = cscale[b,n] * sum_l latT[b,c,l]*Y2[b,l,n]
// M=512 (c, tiled by 64), Ntile=128 (n), K=64 (l).
// ---------------------------------------------------------------------------
__global__ __launch_bounds__(256)
void out_gemm_kernel(const unsigned short* __restrict__ latT,  // (B,512,64) bf16
                     const unsigned short* __restrict__ Y2bf,  // (B,64,6272) bf16
                     const float* __restrict__ cscale,         // (B*6272)
                     float* __restrict__ out)                  // (B,512,6272)
{
  __shared__ unsigned short As[64 * LDSTRIDE];
  __shared__ unsigned short Bs[128 * LDSTRIDE];

  const int t  = threadIdx.x;
  const int n0 = blockIdx.x * 128;
  const int cb = blockIdx.y * 64;
  const int b  = blockIdx.z;

  const int lane  = t & 31;
  const int wave  = t >> 5;
  const int mBase = (wave >> 1) << 4;
  const int nHalf = (wave & 1) << 6;
  v8f acc[4] = {};

  const unsigned short* Ab = latT + (size_t)b * 512 * 64;
  const unsigned short* Yb = Y2bf + (size_t)b * 64 * 6272;

  unsigned       aReg[4];    // paired bf16 copies (K-contig)
  unsigned short bReg[16];   // transposed staging, elementwise
  auto loadTile = [&](int kt) {
    const int l0 = kt * 32;
#pragma unroll
    for (int i = 0; i < 4; ++i) {            // A: latT tile 64(c) x 32(l), 1024 dwords
      int idx = t + i * 256;
      aReg[i] = *(const unsigned*)&Ab[(cb + (idx >> 4)) * 64 + l0 + ((idx & 15) << 1)];
    }
#pragma unroll
    for (int i = 0; i < 16; ++i) {           // B: Y2 tile 32(l) x 128(n)
      int idx = t + i * 256;
      bReg[i] = Yb[(l0 + (idx >> 7)) * 6272 + n0 + (idx & 127)];
    }
  };

  loadTile(0);
  for (int kt = 0; kt < 2; ++kt) {
#pragma unroll
    for (int i = 0; i < 4; ++i) {
      int idx = t + i * 256;
      *(unsigned*)&As[(idx >> 4) * LDSTRIDE + ((idx & 15) << 1)] = aReg[i];
    }
#pragma unroll
    for (int i = 0; i < 16; ++i) {
      int idx = t + i * 256;
      Bs[(idx & 127) * LDSTRIDE + (idx >> 7)] = bReg[i];
    }
    __syncthreads();
    if (kt < 1) loadTile(1);
    Frag a, bf_;
    loadA(a, As, mBase, lane);
#pragma unroll
    for (int j = 0; j < 4; ++j) {
      loadB(bf_, Bs, nHalf + j * 16, lane);
      acc[j] = __builtin_amdgcn_wmma_f32_16x16x32_bf16(false, a.v, false, bf_.v,
                                                       (short)0, acc[j], false, false);
    }
    __syncthreads();
  }

  const int h = lane >> 4, nc = lane & 15;
  const float* cs = cscale + (size_t)b * 6272;
#pragma unroll
  for (int j = 0; j < 4; ++j) {
    int n = n0 + nHalf + j * 16 + nc;
    float sc = cs[n];
#pragma unroll
    for (int r = 0; r < 8; ++r) {
      int c = cb + mBase + r + (h << 3);
      out[((size_t)b * 512 + c) * 6272 + n] = acc[j][r] * sc;
    }
  }
}

// ---------------------------------------------------------------------------

extern "C" void kernel_launch(void* const* d_in, const int* in_sizes, int n_in,
                              void* d_out, int out_size, void* d_ws, size_t ws_size,
                              hipStream_t stream) {
  (void)in_sizes; (void)n_in; (void)out_size; (void)ws_size;

  const float* query   = (const float*)d_in[0];
  const float* key     = (const float*)d_in[1];
  const float* phi_w   = (const float*)d_in[2];
  const float* phi_g   = (const float*)d_in[3];
  const float* phi_b   = (const float*)d_in[4];
  const float* phi_m   = (const float*)d_in[5];
  const float* phi_v   = (const float*)d_in[6];
  const float* phip_w  = (const float*)d_in[7];
  const float* phip_g  = (const float*)d_in[8];
  const float* phip_b  = (const float*)d_in[9];
  const float* phip_m  = (const float*)d_in[10];
  const float* phip_v  = (const float*)d_in[11];
  float* out = (float*)d_out;

  char* ws = (char*)d_ws;
  // Workspace layout (all 256B aligned):
  unsigned short* Y1      = (unsigned short*)(ws);             // 8*64*6272*2   = 6,422,528
  unsigned short* Y2      = (unsigned short*)(ws + 6422528);   //                 6,422,528
  float*          rscale  = (float*)(ws + 12845056);           // 8*64*4        = 2,048
  float*          latent  = (float*)(ws + 12847104);           // 8*64*512*4    = 1,048,576
  unsigned short* latT    = (unsigned short*)(ws + 13895680);  // 8*512*64*2    = 524,288
  float*          cscale  = (float*)(ws + 14419968);           // 8*6272*4      = 200,704
  float*          latPart = (float*)(ws + 14620672);           // 14*8*64*512*4 = 14,680,064
  // total: 29,300,736 bytes

  const dim3 blk(256);

  // 1) visible->latent adjacency (phi on key)
  phi_gemm_kernel<<<dim3(49, 8), blk, 0, stream>>>(key, phi_w, phi_g, phi_b, phi_m, phi_v, Y1);
  // 2) per-(b,l) L2 scales over Nk
  row_scale_kernel<<<dim3(8 * 64), blk, 0, stream>>>(Y1, rscale);
  // 3) latent = (Y1/||row||) @ key^T  (split-K, 448 WGs)
  latent_gemm_kernel<<<dim3(4, 14, 8), blk, 0, stream>>>(Y1, key, latPart);
  latent_reduce_kernel<<<dim3(1024), blk, 0, stream>>>(latPart, rscale, latent);
  // 4) affinity softmax + latent mix, store transposed bf16
  affinity_kernel<<<dim3(8), blk, 0, stream>>>(latent, latT);
  // 5) latent->visible adjacency (phip on query)
  phi_gemm_kernel<<<dim3(49, 8), blk, 0, stream>>>(query, phip_w, phip_g, phip_b, phip_m, phip_v, Y2);
  // 6) per-(b,n) L2 scales over L
  col_scale_kernel<<<dim3((8 * 6272 + 255) / 256), blk, 0, stream>>>(Y2, cscale);
  // 7) out = latent2^T @ (Y2/||col||)
  out_gemm_kernel<<<dim3(49, 8, 8), blk, 0, stream>>>(latT, Y2, cscale, out);
}
